// Encoding_19911468384657
// MI455X (gfx1250) — compile-verified, matrix operand order checked by reading, not run on hardware
//
#include <hip/hip_runtime.h>
#include <hip/hip_bf16.h>

typedef __attribute__((ext_vector_type(16))) __bf16 v16bf;
typedef __attribute__((ext_vector_type(8)))  __bf16 v8bf;
typedef __attribute__((ext_vector_type(4)))  __bf16 v4bf;
typedef __attribute__((ext_vector_type(8)))  float  v8f;

#define B_    16
#define D_    512
#define N_    4096
#define K_    32
#define NCH   16            // n-chunks per batch
#define NC    (N_ / NCH)    // 256 columns per workgroup
#define NT    64            // columns per LDS tile iteration
#define ITERS (NC / NT)     // 4

// LDS strides (bf16 elements / f32 elements). All row strides give 16B-aligned
// rows so b128 fragment loads stay naturally aligned.
#define CB_S  520   // codewords  [32][520]  bf16  (1040B rows, odd-ish bank spread)
#define XN_S  520   // X n-major  [64][520]  bf16
#define XT_S  80    // X d-major  [512][80]  bf16  (160B rows)
#define AK_S  80    // a          [32][80]   bf16
#define XCS_S 33    // xc f32     [64][33]

#define SMEM_BYTES (D_*XT_S*2 + NT*XN_S*2 + K_*CB_S*2 + K_*AK_S*2 \
                    + NT*XCS_S*4 + 16*64*4 + 32*8*4 + 32*4 + 32*4 + 32*4)

// Load one 16-bit WMMA A/B fragment (16 rows x 32 reduction) from LDS.
// Layout per CDNA5 ISA 7.12.2: lane(0..15) -> row, lanes 16..31 same rows,
// element runs [koff + hi*8 .. +7] and [koff + 16 + hi*8 .. +7].
static __device__ inline v16bf frag_ld(const __bf16* row, int koff) {
    const int hi = ((threadIdx.x & 31) >> 4) * 8;
    v8bf lo = *(const v8bf*)(row + koff + hi);
    v8bf hh = *(const v8bf*)(row + koff + 16 + hi);
    return __builtin_shufflevector(lo, hh, 0,1,2,3,4,5,6,7,8,9,10,11,12,13,14,15);
}

extern "C" __global__ __launch_bounds__(256, 1)
void Encoding_19911468384657_kernel(const float* __restrict__ x,
                                    const float* __restrict__ cw,
                                    const float* __restrict__ scale,
                                    float* __restrict__ out) {
    extern __shared__ char smem_raw[];
    __bf16* Xt   = (__bf16*)smem_raw;            // [512][XT_S]
    __bf16* Xn   = Xt + D_ * XT_S;               // [64][XN_S]
    __bf16* Cb   = Xn + NT * XN_S;               // [32][CB_S]
    __bf16* Ak   = Cb + K_ * CB_S;               // [32][AK_S]
    float*  xcs  = (float*)(Ak + K_ * AK_S);     // [64][XCS_S]
    float*  xsqp = xcs + NT * XCS_S;             // [16][64] partial |x|^2
    float*  c2p  = xsqp + 16 * 64;               // [32][8]
    float*  c2   = c2p + 32 * 8;                 // [32]
    float*  s2   = c2 + 32;                      // [32]
    float*  asum = s2 + 32;                      // [32]

    const int tid  = threadIdx.x;
    const int lane = tid & 31;
    const int l15  = lane & 15;
    const int wave = tid >> 5;                   // 0..7
    const int bi    = blockIdx.y;                // batch
    const int chunk = blockIdx.x;                // n-chunk
    const int nbase = chunk * NC;

    // ---------------- init: codewords -> bf16 LDS, c2[k], s2[k] ----------------
    {
        const int krow = tid >> 3;               // 0..31
        const int seg  = tid & 7;                // 0..7 (64 cols each)
        const float* crow = cw + (size_t)krow * D_ + seg * 64;
        float csq = 0.f;
#pragma unroll
        for (int i = 0; i < 16; ++i) {
            float4 v = ((const float4*)crow)[i];
            csq += v.x*v.x + v.y*v.y + v.z*v.z + v.w*v.w;
            v4bf b = { (__bf16)v.x, (__bf16)v.y, (__bf16)v.z, (__bf16)v.w };
            *(v4bf*)(Cb + (size_t)krow * CB_S + seg * 64 + i * 4) = b;
        }
        c2p[krow * 8 + seg] = csq;
        if (tid < K_) asum[tid] = 0.f;
    }
    __syncthreads();
    if (tid < K_) {
        float s = 0.f;
#pragma unroll
        for (int i = 0; i < 8; ++i) s += c2p[tid * 8 + i];
        c2[tid] = s;
        float sc = scale[tid];
        s2[tid] = sc * sc;
    }

    // persistent phase-B accumulators: e^T tiles, wave owns d-range [64*wave, +64)
    v8f accB[4][2];
#pragma unroll
    for (int dt = 0; dt < 4; ++dt)
#pragma unroll
        for (int kt = 0; kt < 2; ++kt)
#pragma unroll
            for (int j = 0; j < 8; ++j) accB[dt][kt][j] = 0.f;

    for (int it = 0; it < ITERS; ++it) {
        __syncthreads();   // previous iteration's readers done before overwrite

        // -------- load X tile (f32 -> bf16, dual layout) + |x|^2 partials -----
        {
            const int dr = tid >> 4;                 // 0..15 (d row group)
            const int c4 = (tid & 15) * 4;           // 0..60 (column base)
            const float* xg = x + (size_t)bi * D_ * N_ + nbase + it * NT + c4;
            float sq0 = 0.f, sq1 = 0.f, sq2 = 0.f, sq3 = 0.f;
#pragma unroll 4
            for (int i = 0; i < 32; ++i) {
                const int d = dr + i * 16;
                float4 v = *(const float4*)(xg + (size_t)d * N_);
                sq0 += v.x*v.x; sq1 += v.y*v.y; sq2 += v.z*v.z; sq3 += v.w*v.w;
                v4bf b = { (__bf16)v.x, (__bf16)v.y, (__bf16)v.z, (__bf16)v.w };
                *(v4bf*)(Xt + (size_t)d * XT_S + c4) = b;   // d-major
                Xn[(size_t)(c4 + 0) * XN_S + d] = b[0];     // n-major
                Xn[(size_t)(c4 + 1) * XN_S + d] = b[1];
                Xn[(size_t)(c4 + 2) * XN_S + d] = b[2];
                Xn[(size_t)(c4 + 3) * XN_S + d] = b[3];
            }
            xsqp[dr * 64 + c4 + 0] = sq0;
            xsqp[dr * 64 + c4 + 1] = sq1;
            xsqp[dr * 64 + c4 + 2] = sq2;
            xsqp[dr * 64 + c4 + 3] = sq3;
        }
        __syncthreads();

        // -------- phase A: xc[k,n] = C(bf16) x X, one 16x16 tile per wave -----
        {
            const int ktile = wave >> 2;             // 0..1
            const int ntile = wave & 3;              // 0..3
            const __bf16* arow = Cb + (size_t)(ktile * 16 + l15) * CB_S;
            const __bf16* brow = Xn + (size_t)(ntile * 16 + l15) * XN_S;
            v8f acc;
#pragma unroll
            for (int j = 0; j < 8; ++j) acc[j] = 0.f;
#pragma unroll
            for (int d0 = 0; d0 < D_; d0 += 32) {
                v16bf a = frag_ld(arow, d0);
                v16bf b = frag_ld(brow, d0);
                acc = __builtin_amdgcn_wmma_f32_16x16x32_bf16(
                          false, a, false, b, (short)0, acc, false, false);
            }
            // C/D layout: value j in lane c -> (M = j + 8*(c>>4), N = c&15)
            const int n  = ntile * 16 + l15;
            const int kb = ktile * 16 + (lane >> 4) * 8;
#pragma unroll
            for (int j = 0; j < 8; ++j) xcs[n * XCS_S + kb + j] = acc[j];
        }
        __syncthreads();

        // -------- softmax over K=32 per column (threads 0..63) ----------------
        if (tid < NT) {
            float x2 = 0.f;
#pragma unroll
            for (int r = 0; r < 16; ++r) x2 += xsqp[r * 64 + tid];
            float sl[K_];
            float mx = -1e30f;
#pragma unroll
            for (int k = 0; k < K_; ++k) {
                float v = s2[k] * (x2 - 2.f * xcs[tid * XCS_S + k] + c2[k]);
                sl[k] = v;
                mx = fmaxf(mx, v);
            }
            float ssum = 0.f;
#pragma unroll
            for (int k = 0; k < K_; ++k) { float e = __expf(sl[k] - mx); sl[k] = e; ssum += e; }
            const float inv = 1.f / ssum;
#pragma unroll
            for (int k = 0; k < K_; ++k) Ak[k * AK_S + tid] = (__bf16)(sl[k] * inv);
        }
        __syncthreads();

        // -------- asum partial (threads 0..31, consistent with bf16 a) --------
        if (tid < K_) {
            float s = 0.f;
#pragma unroll 8
            for (int n = 0; n < NT; ++n) s += (float)Ak[tid * AK_S + n];
            asum[tid] += s;
        }

        // -------- phase B: e^T[d,k] += X[d,n] * a[k,n] ------------------------
        {
            const int d0w = wave * 64;
#pragma unroll
            for (int nn = 0; nn < NT; nn += 32) {
                v16bf b0 = frag_ld(Ak + (size_t)(0 * 16 + l15) * AK_S, nn);
                v16bf b1 = frag_ld(Ak + (size_t)(1 * 16 + l15) * AK_S, nn);
#pragma unroll
                for (int dt = 0; dt < 4; ++dt) {
                    v16bf a = frag_ld(Xt + (size_t)(d0w + dt * 16 + l15) * XT_S, nn);
                    accB[dt][0] = __builtin_amdgcn_wmma_f32_16x16x32_bf16(
                                      false, a, false, b0, (short)0, accB[dt][0], false, false);
                    accB[dt][1] = __builtin_amdgcn_wmma_f32_16x16x32_bf16(
                                      false, a, false, b1, (short)0, accB[dt][1], false, false);
                }
            }
        }
    }
    __syncthreads();   // asum final, all phase-B accumulation done

    // ---------------- emit partial e with folded -asum*C, atomic reduce -------
    {
        const int d0w = wave * 64;
        const int hi8 = (lane >> 4) * 8;
#pragma unroll
        for (int dt = 0; dt < 4; ++dt) {
#pragma unroll
            for (int kt = 0; kt < 2; ++kt) {
                const int k  = kt * 16 + l15;
                const float as = asum[k];
#pragma unroll
                for (int j = 0; j < 8; ++j) {
                    const int d = d0w + dt * 16 + hi8 + j;
                    const float val = accB[dt][kt][j] - as * (float)Cb[(size_t)k * CB_S + d];
                    unsafeAtomicAdd(out + ((size_t)bi * K_ + k) * D_ + d, val);
                }
            }
        }
    }
}

extern "C" void kernel_launch(void* const* d_in, const int* in_sizes, int n_in,
                              void* d_out, int out_size, void* d_ws, size_t ws_size,
                              hipStream_t stream) {
    const float* x  = (const float*)d_in[0];   // [16,512,64,64]
    const float* cw = (const float*)d_in[1];   // [32,512]
    const float* sc = (const float*)d_in[2];   // [32]
    float* out = (float*)d_out;                // [16,32,512]

    (void)in_sizes; (void)n_in; (void)d_ws; (void)ws_size;

    // output is accumulated via atomics -> zero it first (graph-capturable)
    hipMemsetAsync(d_out, 0, (size_t)out_size * sizeof(float), stream);

    hipFuncSetAttribute((const void*)Encoding_19911468384657_kernel,
                        hipFuncAttributeMaxDynamicSharedMemorySize, SMEM_BYTES);

    dim3 grid(NCH, B_);   // 16 n-chunks x 16 batches = 256 workgroups
    Encoding_19911468384657_kernel<<<grid, 256, SMEM_BYTES, stream>>>(x, cw, sc, out);
}
